// LSTM_70205535421125
// MI455X (gfx1250) — matrix operand order repaired and checked by reading, not compiled, not run on hardware
//
#include <hip/hip_runtime.h>
#include <hip/hip_bf16.h>
#include <stdint.h>

// ---------------- problem constants ----------------
#define V_  50257
#define D_  400
#define H_  1024
#define B_  16
#define T_  256
#define DP_ 416            // D padded to multiple of 32 (K dim for WMMA)
#define FH_ 4096           // 4*H
#define M_  4096           // B*T rows

typedef __attribute__((ext_vector_type(16))) __bf16 v16bf;
typedef __attribute__((ext_vector_type(8)))  float  v8f;

union ABf { v16bf v; uint4 q[2]; };   // 32 bytes: one WMMA A/B operand per lane

__device__ __forceinline__ unsigned short f2bf(float f) {
  union { float f; unsigned u; } x; x.f = f;
  unsigned u = x.u;
  unsigned r = u + 0x7FFFu + ((u >> 16) & 1u);   // round-to-nearest-even
  return (unsigned short)(r >> 16);
}

// ---- CDNA5 async global->LDS DMA (ASYNCcnt-tracked; bypasses VGPRs) ----
__device__ __forceinline__ void async_b128(unsigned lds_off, const unsigned short* g) {
  asm volatile("global_load_async_to_lds_b128 %0, %1, off"
               :: "v"(lds_off), "v"((unsigned long long)(uintptr_t)g)
               : "memory");
}
__device__ __forceinline__ void wait_async0() {
  asm volatile("s_wait_asynccnt 0x0" ::: "memory");
}

// ---------------- small utility kernels ----------------
__global__ void k_zero_u32(unsigned* __restrict__ p, long long n) {
  long long i = (long long)blockIdx.x * blockDim.x + threadIdx.x;
  if (i < n) p[i] = 0u;
}

// embedding [V,D] f32 -> [V,DP] bf16 with zero pad
__global__ void k_embed_pad(const float* __restrict__ E, unsigned short* __restrict__ O) {
  long long i = (long long)blockIdx.x * blockDim.x + threadIdx.x;
  long long total = (long long)V_ * DP_;
  if (i >= total) return;
  int row = (int)(i / DP_), col = (int)(i % DP_);
  O[i] = (col < D_) ? f2bf(E[(long long)row * D_ + col]) : (unsigned short)0;
}

// W [K,N] f32 -> Wt [N,Kp] bf16 (transpose + pad K)
__global__ void k_transpose_bf16(const float* __restrict__ W, unsigned short* __restrict__ Wt,
                                 int K, int N, int Kp) {
  long long i = (long long)blockIdx.x * blockDim.x + threadIdx.x;
  long long total = (long long)N * Kp;
  if (i >= total) return;
  int n = (int)(i / Kp), k = (int)(i % Kp);
  Wt[i] = (k < K) ? f2bf(W[(long long)k * N + n]) : (unsigned short)0;
}

// plain f32 -> bf16
__global__ void k_convert_bf16(const float* __restrict__ W, unsigned short* __restrict__ O, long long n) {
  long long i = (long long)blockIdx.x * blockDim.x + threadIdx.x;
  if (i < n) O[i] = f2bf(W[i]);
}

// gather token rows out of padded bf16 table
__global__ void k_gather(const int* __restrict__ x, const unsigned short* __restrict__ tab,
                         unsigned short* __restrict__ out) {
  long long i = (long long)blockIdx.x * blockDim.x + threadIdx.x;
  long long total = (long long)M_ * DP_;
  if (i >= total) return;
  int row = (int)(i / DP_), col = (int)(i % DP_);
  out[i] = tab[(long long)x[row] * DP_ + col];
}

// ---------------- WMMA GEMM: C[M,N] = A[M,K] * Bt[N,K]^T (+bias) ----------------
// Block = 8 waves = 128 rows x 64 cols. A and B tiles double-buffered in LDS,
// filled by global_load_async_to_lds_b128 (ASYNCcnt). Padded row stride of 40
// halfs (80 B) keeps ds_load_b128 16B-aligned and bank-conflict-free.
// K multiple of 32; M multiple of 128 (all call sites); N arbitrary
// (stage source rows clamped, stores masked).
#define SBSTRIDE 40                       // halfs per staged row (32 data + 8 pad)
#define SB_BYTES (64  * SBSTRIDE * 2)     // one B buffer: 5120 B
#define SA_BYTES (128 * SBSTRIDE * 2)     // one A buffer: 10240 B

__global__ __launch_bounds__(256) void k_gemm_wmma(
    const unsigned short* __restrict__ A, int lda,
    const unsigned short* __restrict__ Bt, int ldb,
    const float* __restrict__ bias,
    float* __restrict__ Cf, long long ldc,
    unsigned short* __restrict__ Cb, int ldcb,
    int M, int N, int K)
{
  __shared__ alignas(16) unsigned short sA[2][128 * SBSTRIDE];
  __shared__ alignas(16) unsigned short sB[2][64 * SBSTRIDE];

  const int tid  = threadIdx.x;
  const int lane = tid & 31;
  const int wave = tid >> 5;
  const int rowBase = blockIdx.y * 128;
  const int rowTile = rowBase + wave * 16;
  const int n0 = blockIdx.x * 64;
  const int lm = lane & 15;
  const int hi = lane >> 4;

  // ---- staging: 256 threads; B = 256 x b128, A = 512 x b128 (2 per thread)
  const int srow = tid >> 2;              // 0..63
  const int sseg = tid & 3;               // 0..3 (8 halfs each)
  int scol = n0 + srow;
  int scl  = scol < N ? scol : N - 1;
  const unsigned short* bsrc  = Bt + (long long)scl * ldb + sseg * 8;
  const unsigned short* asrc0 = A + (long long)(rowBase + srow)      * lda + sseg * 8;
  const unsigned short* asrc1 = A + (long long)(rowBase + 64 + srow) * lda + sseg * 8;

  const unsigned sA_lds = (unsigned)(unsigned long long)(uintptr_t)&sA[0][0];
  const unsigned sB_lds = (unsigned)(unsigned long long)(uintptr_t)&sB[0][0];
  const unsigned bDst  = sB_lds + (unsigned)(srow * SBSTRIDE + sseg * 8) * 2;
  const unsigned aDst0 = sA_lds + (unsigned)(srow * SBSTRIDE + sseg * 8) * 2;
  const unsigned aDst1 = sA_lds + (unsigned)((64 + srow) * SBSTRIDE + sseg * 8) * 2;

  // ---- per-lane LDS read bases (buffer 0); buffer 1 = +buffer size
  const unsigned short* aRd = &sA[0][(wave * 16 + lm) * SBSTRIDE + hi * 16];
  const unsigned short* bRd = &sB[0][lm * SBSTRIDE + hi * 16];

  v8f acc[4];
#pragma unroll
  for (int j = 0; j < 4; ++j) {
    int n = n0 + j * 16 + lm;
    int nc = n < N ? n : N - 1;
    float bv = bias ? bias[nc] : 0.0f;
#pragma unroll
    for (int r = 0; r < 8; ++r) acc[j][r] = bv;
  }

  // prologue: async-stage chunk 0 into buffer 0
  async_b128(bDst,  bsrc);
  async_b128(aDst0, asrc0);
  async_b128(aDst1, asrc1);
  wait_async0();
  __syncthreads();

  int buf = 0;
  for (int kc = 0; kc < K; kc += 32) {
    const int nbuf = buf ^ 1;
    if (kc + 32 < K) {                    // async-prefetch next chunk
      async_b128(bDst  + (unsigned)nbuf * SB_BYTES, bsrc  + kc + 32);
      async_b128(aDst0 + (unsigned)nbuf * SA_BYTES, asrc0 + kc + 32);
      async_b128(aDst1 + (unsigned)nbuf * SA_BYTES, asrc1 + kc + 32);
    }
    const unsigned short* ap  = aRd + buf * (128 * SBSTRIDE);
    const unsigned short* bpb = bRd + buf * (64 * SBSTRIDE);
    ABf a;
    a.q[0] = *(const uint4*)(ap);
    a.q[1] = *(const uint4*)(ap + 8);
#pragma unroll
    for (int j = 0; j < 4; ++j) {
      const unsigned short* bp = bpb + j * (16 * SBSTRIDE);
      ABf b;
      b.q[0] = *(const uint4*)(bp);
      b.q[1] = *(const uint4*)(bp + 8);
      acc[j] = __builtin_amdgcn_wmma_f32_16x16x32_bf16(
          false, a.v, false, b.v, (short)0, acc[j], false, false);
    }
    wait_async0();                        // next chunk landed in LDS
    __syncthreads();
    buf = nbuf;
  }

#pragma unroll
  for (int j = 0; j < 4; ++j) {
#pragma unroll
    for (int r = 0; r < 8; ++r) {
      int m = rowTile + r + hi * 8;
      int n = n0 + j * 16 + lm;
      if (n < N) {
        float v = acc[j][r];
        if (Cf) Cf[(long long)m * ldc + n] = v;
        if (Cb) Cb[(long long)m * ldcb + n] = f2bf(v);
      }
    }
  }
}

// ---------------- one LSTM timestep ----------------
// One wave per block; block j owns H-columns [16j,16j+16): computes the four
// 16x16 gate tiles (acc preloaded from xp), register-double-buffered K loop,
// then fp32 gate math + cell update.
__global__ __launch_bounds__(32) void k_lstm_step(
    const float* __restrict__ xp,            // [B,T,4H], row = b*T+t
    const unsigned short* __restrict__ h_in, // [B,H] bf16 (previous h)
    const unsigned short* __restrict__ whhT, // [4H,H] bf16 (W_hh transposed)
    float* __restrict__ c,                   // [B,H] fp32 cell state (in/out)
    float* __restrict__ hf,                  // [B,H] fp32 h (final-state copy)
    unsigned short* __restrict__ h_out,      // [B,H] bf16 next h (!= h_in)
    unsigned short* __restrict__ y,          // [B*T,H] bf16 layer output
    int t)
{
  const int lane = threadIdx.x & 31;
  const int lm = lane & 15;
  const int hi = lane >> 4;
  const int jc = blockIdx.x * 16;
  const int col = jc + lm;
  const int aoff = hi * 16;
  const int boff = hi * 16;

  const unsigned short* abase = h_in + lm * H_ + aoff;
  const unsigned short* bb0 = whhT + (long long)(0 * H_ + col) * H_ + boff;
  const unsigned short* bb1 = whhT + (long long)(1 * H_ + col) * H_ + boff;
  const unsigned short* bb2 = whhT + (long long)(2 * H_ + col) * H_ + boff;
  const unsigned short* bb3 = whhT + (long long)(3 * H_ + col) * H_ + boff;

  v8f acc[4];
#pragma unroll
  for (int q = 0; q < 4; ++q) {
#pragma unroll
    for (int r = 0; r < 8; ++r) {
      int m = r + hi * 8;                       // batch index
      acc[q][r] = xp[(long long)(m * T_ + t) * FH_ + q * H_ + col];
    }
  }

#define LD_CHUNK(kc_, a_, b_)                                                 \
  {                                                                           \
    (a_).q[0] = *(const uint4*)(abase + (kc_));                               \
    (a_).q[1] = *(const uint4*)(abase + (kc_) + 8);                           \
    (b_)[0].q[0] = *(const uint4*)(bb0 + (kc_));                              \
    (b_)[0].q[1] = *(const uint4*)(bb0 + (kc_) + 8);                          \
    (b_)[1].q[0] = *(const uint4*)(bb1 + (kc_));                              \
    (b_)[1].q[1] = *(const uint4*)(bb1 + (kc_) + 8);                          \
    (b_)[2].q[0] = *(const uint4*)(bb2 + (kc_));                              \
    (b_)[2].q[1] = *(const uint4*)(bb2 + (kc_) + 8);                          \
    (b_)[3].q[0] = *(const uint4*)(bb3 + (kc_));                              \
    (b_)[3].q[1] = *(const uint4*)(bb3 + (kc_) + 8);                          \
  }

#define DO_WMMA(a_, b_)                                                       \
  {                                                                           \
    _Pragma("unroll")                                                         \
    for (int q = 0; q < 4; ++q)                                               \
      acc[q] = __builtin_amdgcn_wmma_f32_16x16x32_bf16(                       \
          false, (a_).v, false, (b_)[q].v, (short)0, acc[q], false, false);   \
  }

  ABf aA, aB, bA[4], bB[4];
  LD_CHUNK(0, aA, bA);
#pragma unroll 1
  for (int kc = 0; kc < H_; kc += 64) {
    if (kc + 32 < H_) LD_CHUNK(kc + 32, aB, bB);   // prefetch while computing
    DO_WMMA(aA, bA);
    if (kc + 64 < H_) LD_CHUNK(kc + 64, aA, bA);
    DO_WMMA(aB, bB);
  }
#undef LD_CHUNK
#undef DO_WMMA

#pragma unroll
  for (int r = 0; r < 8; ++r) {
    int m = r + hi * 8;
    float iv = 1.0f / (1.0f + __expf(-acc[0][r]));
    float fv = 1.0f / (1.0f + __expf(-acc[1][r]));
    float gv = tanhf(acc[2][r]);
    float ov = 1.0f / (1.0f + __expf(-acc[3][r]));
    int ci = m * H_ + col;
    float cn = fv * c[ci] + iv * gv;
    float hn = ov * tanhf(cn);
    c[ci] = cn;
    hf[ci] = hn;
    unsigned short hb = f2bf(hn);
    h_out[ci] = hb;
    y[(long long)(m * T_ + t) * H_ + col] = hb;
  }
}

// ---------------- host orchestration ----------------
static inline dim3 g1(long long total, int thr) { return dim3((unsigned)((total + thr - 1) / thr)); }

extern "C" void kernel_launch(void* const* d_in, const int* in_sizes, int n_in,
                              void* d_out, int out_size, void* d_ws, size_t ws_size,
                              hipStream_t stream) {
  (void)in_sizes; (void)n_in; (void)out_size; (void)ws_size;

  const int*   x    = (const int*)d_in[0];
  const float* E    = (const float*)d_in[1];
  const float* wih0 = (const float*)d_in[2];
  const float* whh0 = (const float*)d_in[3];
  const float* b0   = (const float*)d_in[4];
  const float* wih1 = (const float*)d_in[5];
  const float* whh1 = (const float*)d_in[6];
  const float* b1   = (const float*)d_in[7];
  const float* pw   = (const float*)d_in[8];
  const float* fcb  = (const float*)d_in[9];
  float* out = (float*)d_out;

  char* p = (char*)d_ws;
  auto alloc = [&](size_t bytes) -> void* {
    void* r = (void*)p;
    p += (bytes + 255) & ~(size_t)255;
    return r;
  };
  unsigned short* embT   = (unsigned short*)alloc((size_t)V_ * DP_ * 2);
  unsigned short* wih0T  = (unsigned short*)alloc((size_t)FH_ * DP_ * 2);
  unsigned short* whh0T  = (unsigned short*)alloc((size_t)FH_ * H_ * 2);
  unsigned short* wih1T  = (unsigned short*)alloc((size_t)FH_ * H_ * 2);
  unsigned short* whh1T  = (unsigned short*)alloc((size_t)FH_ * H_ * 2);
  unsigned short* pwT    = (unsigned short*)alloc((size_t)D_ * H_ * 2);
  unsigned short* embSeq = (unsigned short*)alloc((size_t)M_ * DP_ * 2);
  float*          xp     = (float*)alloc((size_t)M_ * FH_ * 4);
  unsigned short* y1     = (unsigned short*)alloc((size_t)M_ * H_ * 2);
  unsigned short* y2     = (unsigned short*)alloc((size_t)M_ * H_ * 2);
  unsigned short* projP  = (unsigned short*)alloc((size_t)M_ * DP_ * 2);
  unsigned short* hA     = (unsigned short*)alloc((size_t)B_ * H_ * 2);
  unsigned short* hB     = (unsigned short*)alloc((size_t)B_ * H_ * 2);
  float*          c0     = (float*)alloc((size_t)B_ * H_ * 4);
  float*          h0f    = (float*)alloc((size_t)B_ * H_ * 4);
  float*          c1b    = (float*)alloc((size_t)B_ * H_ * 4);
  float*          h1f    = (float*)alloc((size_t)B_ * H_ * 4);

  const int thr = 256;

  k_embed_pad<<<g1((long long)V_ * DP_, thr), thr, 0, stream>>>(E, embT);
  k_transpose_bf16<<<g1((long long)FH_ * DP_, thr), thr, 0, stream>>>(wih0, wih0T, D_, FH_, DP_);
  k_transpose_bf16<<<g1((long long)FH_ * H_, thr), thr, 0, stream>>>(whh0, whh0T, H_, FH_, H_);
  k_transpose_bf16<<<g1((long long)FH_ * H_, thr), thr, 0, stream>>>(wih1, wih1T, H_, FH_, H_);
  k_transpose_bf16<<<g1((long long)FH_ * H_, thr), thr, 0, stream>>>(whh1, whh1T, H_, FH_, H_);
  k_convert_bf16<<<g1((long long)D_ * H_, thr), thr, 0, stream>>>(pw, pwT, (long long)D_ * H_);

  k_gather<<<g1((long long)M_ * DP_, thr), thr, 0, stream>>>(x, embT, embSeq);

  // -------- layer 0: xp0 = emb @ W_ih0 + b0 --------
  k_gemm_wmma<<<dim3(FH_ / 64, M_ / 128), 256, 0, stream>>>(
      embSeq, DP_, wih0T, DP_, b0, xp, (long long)FH_, (unsigned short*)nullptr, 0, M_, FH_, DP_);

  k_zero_u32<<<g1((long long)B_ * H_ / 2, thr), thr, 0, stream>>>((unsigned*)hA, (long long)B_ * H_ / 2);
  k_zero_u32<<<g1((long long)B_ * H_, thr), thr, 0, stream>>>((unsigned*)c0, (long long)B_ * H_);
  {
    unsigned short* hin = hA; unsigned short* hout = hB;
    for (int t = 0; t < T_; ++t) {
      k_lstm_step<<<H_ / 16, 32, 0, stream>>>(xp, hin, whh0T, c0, h0f, hout, y1, t);
      unsigned short* tmp = hin; hin = hout; hout = tmp;
    }
  }

  // -------- layer 1: xp1 = y1 @ W_ih1 + b1 --------
  k_gemm_wmma<<<dim3(FH_ / 64, M_ / 128), 256, 0, stream>>>(
      y1, H_, wih1T, H_, b1, xp, (long long)FH_, (unsigned short*)nullptr, 0, M_, FH_, H_);

  k_zero_u32<<<g1((long long)B_ * H_ / 2, thr), thr, 0, stream>>>((unsigned*)hA, (long long)B_ * H_ / 2);
  k_zero_u32<<<g1((long long)B_ * H_, thr), thr, 0, stream>>>((unsigned*)c1b, (long long)B_ * H_);
  {
    unsigned short* hin = hA; unsigned short* hout = hB;
    for (int t = 0; t < T_; ++t) {
      k_lstm_step<<<H_ / 16, 32, 0, stream>>>(xp, hin, whh1T, c1b, h1f, hout, y2, t);
      unsigned short* tmp = hin; hin = hout; hout = tmp;
    }
  }

  // -------- projection: proj = y2 @ proj_w^T (bf16 out, K-padded) --------
  k_zero_u32<<<g1((long long)M_ * DP_ / 2, thr), thr, 0, stream>>>((unsigned*)projP, (long long)M_ * DP_ / 2);
  k_gemm_wmma<<<dim3((D_ + 63) / 64, M_ / 128), 256, 0, stream>>>(
      y2, H_, pwT, H_, (const float*)nullptr, (float*)nullptr, 0, projP, DP_, M_, D_, H_);

  // -------- logits = proj @ E^T + fc_b (into d_out) --------
  k_gemm_wmma<<<dim3((V_ + 63) / 64, M_ / 128), 256, 0, stream>>>(
      projP, DP_, embT, DP_, fcb, out, (long long)V_, (unsigned short*)nullptr, 0, M_, V_, DP_);

  // -------- final states h1,c1,h2,c2 --------
  float* tail = out + (long long)M_ * V_;
  hipMemcpyAsync(tail + 0 * (size_t)B_ * H_, h0f, (size_t)B_ * H_ * 4, hipMemcpyDeviceToDevice, stream);
  hipMemcpyAsync(tail + 1 * (size_t)B_ * H_, c0,  (size_t)B_ * H_ * 4, hipMemcpyDeviceToDevice, stream);
  hipMemcpyAsync(tail + 2 * (size_t)B_ * H_, h1f, (size_t)B_ * H_ * 4, hipMemcpyDeviceToDevice, stream);
  hipMemcpyAsync(tail + 3 * (size_t)B_ * H_, c1b, (size_t)B_ * H_ * 4, hipMemcpyDeviceToDevice, stream);
}
